// MultiHeadAttention_7816840478980
// MI455X (gfx1250) — matrix operand order repaired and checked by reading, not compiled, not run on hardware
//
#include <hip/hip_runtime.h>
#include <hip/hip_bf16.h>

#define DEV static __device__ __forceinline__

typedef __bf16 bf16_t;
typedef __attribute__((ext_vector_type(16))) __bf16 bf16x16;
typedef __attribute__((ext_vector_type(8)))  __bf16 bf16x8;
typedef __attribute__((ext_vector_type(8)))  float   f32x8;

constexpr int Bb = 4, Tt = 1024, Ss = 1024, E = 1024, H = 16, D = 64;
constexpr int ROWS = Bb * Tt;           // 4096 rows for all big GEMMs

DEV f32x8 wmma_bf16(bf16x16 a, bf16x16 b, f32x8 c) {
    // D = A(16x32) * B(32x16) + C, f32 accumulate
    return __builtin_amdgcn_wmma_f32_16x16x32_bf16(false, a, false, b,
                                                   (short)0, c, false, false);
}

DEV f32x8 zero8() {
    f32x8 v;
    #pragma unroll
    for (int j = 0; j < 8; ++j) v[j] = 0.0f;
    return v;
}

DEV bf16x16 join8(bf16x8 lo, bf16x8 hi) {
    bf16x16 r;
    #pragma unroll
    for (int j = 0; j < 8; ++j) { r[j] = lo[j]; r[j + 8] = hi[j]; }
    return r;
}

// ---------------------------------------------------------------------------
// Generic 4096x1024x1024 GEMM, bf16 WMMA, block tile 128x128 (8 waves, 32x64
// per wave). Weight W (fp32 [K,N] row-major) is repacked through LDS into the
// B-fragment layout [n][khalf][16 k: 8h+0..7, then 16+8h+0..7].
// OUT_MODE 0: bf16 row-major (scale applied)       -> Q, K
// OUT_MODE 1: bf16 transposed V layout [bh*64+d][s] -> V
// OUT_MODE 2: fp32 row-major + bias                 -> final out
// ---------------------------------------------------------------------------
template <bool A_BF16, int OUT_MODE>
__global__ __launch_bounds__(256)
void gemm_k(const void* __restrict__ Aptr, const float* __restrict__ W,
            void* __restrict__ Out, const float* __restrict__ bias, float scale)
{
    __shared__ bf16_t ldsB[128 * 32];     // [n][kh][16] : idx = n*32 + ((k>>3)&1)*16 + (k>>4)*8 + (k&7)

    const int nb = blockIdx.x;            // N/128 = 8
    const int mb = blockIdx.y;            // M/128 = 32
    const int lane = threadIdx.x & 31;
    const int wave = threadIdx.x >> 5;
    const int wm = wave & 3;              // 4 waves along M
    const int wn = wave >> 2;             // 2 waves along N
    const int m  = lane & 15;
    const int kh = lane >> 4;

    f32x8 acc[2][4];
    #pragma unroll
    for (int i = 0; i < 2; ++i)
        #pragma unroll
        for (int j = 0; j < 4; ++j) acc[i][j] = zero8();

    const int fk  = threadIdx.x >> 3;        // 0..31  (k within chunk)
    const int fn0 = (threadIdx.x & 7) * 16;  // 0..112 (n start)

    for (int kc = 0; kc < E / 32; ++kc) {
        __syncthreads();
        // ---- repack 32x128 fp32 weight chunk into B-fragment LDS layout ----
        {
            const float* wp = W + (size_t)(kc * 32 + fk) * E + nb * 128 + fn0;
            union { float4 q[4]; float f[16]; } u;
            u.q[0] = ((const float4*)wp)[0];
            u.q[1] = ((const float4*)wp)[1];
            u.q[2] = ((const float4*)wp)[2];
            u.q[3] = ((const float4*)wp)[3];
            const int koff = ((fk >> 3) & 1) * 16 + (fk >> 4) * 8 + (fk & 7);
            #pragma unroll
            for (int i = 0; i < 16; ++i)
                ldsB[(fn0 + i) * 32 + koff] = (bf16_t)u.f[i];
        }
        __syncthreads();

        // ---- A fragments, straight from global (L2-resident) ----
        bf16x16 afr[2];
        #pragma unroll
        for (int mt = 0; mt < 2; ++mt) {
            const size_t row = (size_t)(mb * 128 + wm * 32 + mt * 16 + m);
            if constexpr (!A_BF16) {
                const float* ap = (const float*)Aptr + row * E + kc * 32 + kh * 8;
                union { float4 q[2]; float f[8]; } lo, hi;
                lo.q[0] = ((const float4*)ap)[0];
                lo.q[1] = ((const float4*)ap)[1];
                hi.q[0] = *(const float4*)(ap + 16);
                hi.q[1] = *(const float4*)(ap + 20);
                bf16x16 a;
                #pragma unroll
                for (int j = 0; j < 8; ++j) {
                    a[j]     = (bf16_t)lo.f[j];
                    a[j + 8] = (bf16_t)hi.f[j];
                }
                afr[mt] = a;
            } else {
                const bf16_t* ap = (const bf16_t*)Aptr + row * E + kc * 32 + kh * 8;
                afr[mt] = join8(*(const bf16x8*)ap, *(const bf16x8*)(ap + 16));
            }
        }
        // ---- B fragments from LDS + WMMA ----
        #pragma unroll
        for (int nt = 0; nt < 4; ++nt) {
            const int nn = wn * 64 + nt * 16 + m;
            bf16x16 bfr = *(const bf16x16*)&ldsB[nn * 32 + kh * 16];
            acc[0][nt] = wmma_bf16(afr[0], bfr, acc[0][nt]);
            acc[1][nt] = wmma_bf16(afr[1], bfr, acc[1][nt]);
        }
    }

    // ---- epilogue ----
    #pragma unroll
    for (int mt = 0; mt < 2; ++mt) {
        #pragma unroll
        for (int nt = 0; nt < 4; ++nt) {
            const int col  = nb * 128 + wn * 64 + nt * 16 + m;
            const int row0 = mb * 128 + wm * 32 + mt * 16 + 8 * kh;
            if constexpr (OUT_MODE == 0) {
                bf16_t* o = (bf16_t*)Out;
                #pragma unroll
                for (int r = 0; r < 8; ++r)
                    o[(size_t)(row0 + r) * E + col] = (bf16_t)(acc[mt][nt][r] * scale);
            } else if constexpr (OUT_MODE == 1) {
                const int bb = row0 >> 10;          // batch
                const int sb = row0 & 1023;         // seq pos (mult of 8)
                const int hc = col >> 6, dc = col & 63;
                bf16x8 v;
                #pragma unroll
                for (int r = 0; r < 8; ++r) v[r] = (bf16_t)acc[mt][nt][r];
                *(bf16x8*)((bf16_t*)Out + ((size_t)(bb * H + hc) * D + dc) * Ss + sb) = v;
            } else {
                float* o = (float*)Out;
                const float bi = bias[col];
                #pragma unroll
                for (int r = 0; r < 8; ++r)
                    o[(size_t)(row0 + r) * E + col] = acc[mt][nt][r] + bi;
            }
        }
    }
    (void)scale;
}

// ---------------------------------------------------------------------------
// Attention: grid (B*H, T/128). 8 waves, 16 t-rows each.
// Phase 1: scores = mask + QK^T (WMMA), raw scores -> attn buffer, row max.
// Phase 2: exp-sum (L2-hot re-read), then normalize + write attn fp32 and
//          fused P*V WMMA (P transposed D->A layout through per-wave LDS).
// ---------------------------------------------------------------------------
__global__ __launch_bounds__(256)
void attn_k(const bf16_t* __restrict__ Qbf, const bf16_t* __restrict__ Kbf,
            const bf16_t* __restrict__ Vt,  const float* __restrict__ spm,
            const float* __restrict__ tpm,  const float* __restrict__ am,
            const int* __restrict__ iscp,   float* __restrict__ attn,
            bf16_t* __restrict__ Obf)
{
    __shared__ bf16_t ldsP[8 * 16 * 32];     // per-wave 16x32 bf16 transpose tile

    const int bh = blockIdx.x;               // b*H + h
    const int tb = blockIdx.y;
    const int b = bh >> 4, h = bh & 15;
    const int lane = threadIdx.x & 31;
    const int wave = threadIdx.x >> 5;
    const int t0 = tb * 128 + wave * 16;
    const int m  = lane & 15;
    const int kh = lane >> 4;
    const int isc = *iscp;
    bf16_t* myLds = &ldsP[wave * 16 * 32];

    // Q A-fragments (two 32-wide d chunks), q already scaled by 1/sqrt(D)
    bf16x16 aq[2];
    #pragma unroll
    for (int c = 0; c < 2; ++c) {
        const bf16_t* qp = Qbf + (size_t)(b * Tt + t0 + m) * E + h * 64 + c * 32 + kh * 16;
        aq[c] = join8(*(const bf16x8*)qp, *(const bf16x8*)(qp + 8));
    }

    int   trow[8];  float tpv[8];  float rmax[8];
    #pragma unroll
    for (int r = 0; r < 8; ++r) {
        trow[r] = t0 + r + 8 * kh;
        tpv[r]  = tpm[b * Tt + trow[r]];
        rmax[r] = -3.4e38f;
    }
    const size_t attnBase = (size_t)bh * Tt * Ss;

    // ---- phase 1: scores + mask, raw to global, track max ----
    for (int st = 0; st < Ss / 16; ++st) {
        bf16x16 bk[2];
        #pragma unroll
        for (int c = 0; c < 2; ++c) {
            const bf16_t* kp = Kbf + (size_t)(b * Ss + st * 16 + m) * E
                             + h * 64 + c * 32 + kh * 16;
            bk[c] = *(const bf16x16*)kp;
        }
        f32x8 d = zero8();
        d = wmma_bf16(aq[0], bk[0], d);
        d = wmma_bf16(aq[1], bk[1], d);

        const int sCol = st * 16 + m;
        const float spv = spm[b * Ss + sCol];
        #pragma unroll
        for (int r = 0; r < 8; ++r) {
            const int t = trow[r];
            float mv = tpv[r] * spv * am[(size_t)(b * Tt + t) * Ss + sCol];
            if (isc && (sCol > t)) mv = 0.0f;
            const float v = d[r] + mv;
            attn[attnBase + (size_t)t * Ss + sCol] = v;
            rmax[r] = fmaxf(rmax[r], v);
        }
    }
    #pragma unroll
    for (int off = 1; off < 16; off <<= 1)
        #pragma unroll
        for (int r = 0; r < 8; ++r)
            rmax[r] = fmaxf(rmax[r], __shfl_xor(rmax[r], off, 16));

    // ---- phase 2a: exp sums (re-read raw scores from L2) ----
    float rsum[8];
    #pragma unroll
    for (int r = 0; r < 8; ++r) rsum[r] = 0.0f;
    for (int st = 0; st < Ss / 16; ++st) {
        const int sCol = st * 16 + m;
        #pragma unroll
        for (int r = 0; r < 8; ++r)
            rsum[r] += __expf(attn[attnBase + (size_t)trow[r] * Ss + sCol] - rmax[r]);
    }
    #pragma unroll
    for (int off = 1; off < 16; off <<= 1)
        #pragma unroll
        for (int r = 0; r < 8; ++r)
            rsum[r] += __shfl_xor(rsum[r], off, 16);
    float rinv[8];
    #pragma unroll
    for (int r = 0; r < 8; ++r) rinv[r] = 1.0f / rsum[r];

    // ---- phase 2b: normalize, write attn, fused P*V ----
    f32x8 o[4];
    #pragma unroll
    for (int nt = 0; nt < 4; ++nt) o[nt] = zero8();

    for (int sc = 0; sc < Ss / 32; ++sc) {
        const int s0 = sc * 32;
        #pragma unroll
        for (int half = 0; half < 2; ++half) {
            const int sCol = s0 + half * 16 + m;
            #pragma unroll
            for (int r = 0; r < 8; ++r) {
                const size_t ai = attnBase + (size_t)trow[r] * Ss + sCol;
                const float p = __expf(attn[ai] - rmax[r]) * rinv[r];
                attn[ai] = p;                                    // final attn fp32
                myLds[(r + 8 * kh) * 32 + half * 16 + m] = (bf16_t)p;
            }
        }
        asm volatile("s_wait_dscnt 0" ::: "memory");             // D->A transpose ready
        bf16x16 ap = join8(*(const bf16x8*)&myLds[m * 32 + kh * 16],
                           *(const bf16x8*)&myLds[m * 32 + kh * 16 + 8]);
        #pragma unroll
        for (int nt = 0; nt < 4; ++nt) {
            const bf16_t* vp = Vt + (size_t)(bh * 64 + nt * 16 + m) * Ss + s0 + kh * 16;
            bf16x16 bv = *(const bf16x16*)vp;
            o[nt] = wmma_bf16(ap, bv, o[nt]);
        }
    }

    // ---- store O in [T, B, E] (concat heads) as bf16 for final projection ----
    #pragma unroll
    for (int nt = 0; nt < 4; ++nt) {
        const int dcol = h * 64 + nt * 16 + m;
        #pragma unroll
        for (int r = 0; r < 8; ++r)
            Obf[(size_t)(trow[r] * Bb + b) * E + dcol] = (bf16_t)o[nt][r];
    }
}

// ---------------------------------------------------------------------------
extern "C" void kernel_launch(void* const* d_in, const int* in_sizes, int n_in,
                              void* d_out, int out_size, void* d_ws, size_t ws_size,
                              hipStream_t stream) {
    const float* src = (const float*)d_in[0];   // source_emb [B,S,E]
    const float* tgt = (const float*)d_in[1];   // target_emb [B,T,E]
    const float* spm = (const float*)d_in[2];
    const float* tpm = (const float*)d_in[3];
    const float* am  = (const float*)d_in[4];
    const int*   isc = (const int*)d_in[5];
    const float* Wq  = (const float*)d_in[6];
    const float* Wk  = (const float*)d_in[7];
    const float* Wv  = (const float*)d_in[8];
    const float* Wo  = (const float*)d_in[9];
    const float* bo  = (const float*)d_in[10];

    float* out  = (float*)d_out;                       // [T,B,E]
    float* attn = out + (size_t)Tt * Bb * E;           // [B,H,T,S]

    bf16_t* Qbf = (bf16_t*)d_ws;                       // 4 x 8MB in workspace
    bf16_t* Kbf = Qbf + (size_t)ROWS * E;
    bf16_t* Vt  = Kbf + (size_t)ROWS * E;
    bf16_t* Obf = Vt  + (size_t)ROWS * E;

    dim3 blk(256);
    dim3 gG(E / 128, ROWS / 128);                      // (8, 32)
    gemm_k<false, 0><<<gG, blk, 0, stream>>>(tgt, Wq, Qbf, nullptr, 0.125f);
    gemm_k<false, 0><<<gG, blk, 0, stream>>>(src, Wk, Kbf, nullptr, 1.0f);
    gemm_k<false, 1><<<gG, blk, 0, stream>>>(src, Wv, Vt,  nullptr, 1.0f);

    attn_k<<<dim3(Bb * H, Tt / 128), blk, 0, stream>>>(Qbf, Kbf, Vt, spm, tpm,
                                                       am, isc, attn, Obf);

    gemm_k<true, 2><<<gG, blk, 0, stream>>>(Obf, Wo, out, bo, 1.0f);
    (void)in_sizes; (void)n_in; (void)out_size; (void)ws_size;
}
